// LuminaLM_25417616458114
// MI455X (gfx1250) — compile-verified, matrix operand order checked by reading, not run on hardware
//
#include <hip/hip_runtime.h>
#include <hip/hip_bf16.h>

// ---------------------------------------------------------------------------
// GPT-2 forward for MI455X (gfx1250).
//   - All GEMMs: bf16 WMMA (v_wmma_f32_16x16x32_bf16), f32 accumulate.
//   - Activations feeding GEMM A-sides are stored bf16 in memory, so A tiles
//     stage into LDS via GLOBAL_LOAD_ASYNC_TO_LDS_B128 (ASYNCcnt path); W
//     tiles convert f32->bf16 in flight (co-executes with WMMA).
//   - B=2 T=1024 D=1024 L=4 H=16 V=32000: every GEMM dim is a multiple of
//     (128,128,64), so no bounds checks.
// ---------------------------------------------------------------------------

typedef __attribute__((ext_vector_type(16))) __bf16 v16bf;
typedef __attribute__((ext_vector_type(8)))  float  v8f;
typedef unsigned short u16;

// Types matching the async-LDS builtin prototype reported by hipcc:
//   param0: int __vector_size(16) __device__(AS1) *
typedef int v4i_vs __attribute__((vector_size(16)));
typedef __attribute__((address_space(1))) v4i_vs* gptr128;
typedef __attribute__((address_space(3))) v4i_vs* lptr128;

#define TM 128
#define TN 128
#define TK 64
#define LDS_PAD 8   // rows of (TK+LDS_PAD) u16 = 144 B: 16B-aligned, 36-dword stride

#if defined(__has_builtin)
#if __has_builtin(__builtin_amdgcn_global_load_async_to_lds_b128)
#define HAVE_ASYNC_LDS 1
#endif
#endif
#ifndef HAVE_ASYNC_LDS
#define HAVE_ASYNC_LDS 0
#endif

__device__ __forceinline__ u16 f32_to_bf16(float f) {
    union { float f; unsigned u; } c; c.f = f;
    unsigned u = c.u;
    u += 0x7FFFu + ((u >> 16) & 1u);    // round-to-nearest-even
    return (u16)(u >> 16);
}

union FragBF16 { v16bf v; unsigned u[8]; };

__device__ __forceinline__ void load_frag(const u16* p, FragBF16& f) {
    // 8 bf16-pairs: K = {0,2,4,6} and {16,18,20,22} (+half handled by caller)
    const uint4 lo = *(const uint4*)p;
    const uint4 hi = *(const uint4*)(p + 16);
    f.u[0] = lo.x; f.u[1] = lo.y; f.u[2] = lo.z; f.u[3] = lo.w;
    f.u[4] = hi.x; f.u[5] = hi.y; f.u[6] = hi.z; f.u[7] = hi.w;
}

__device__ __forceinline__ void async_copy16(const u16* gsrc, u16* ldst) {
#if HAVE_ASYNC_LDS
    u16* s = const_cast<u16*>(gsrc);
    __builtin_amdgcn_global_load_async_to_lds_b128((gptr128)s, (lptr128)ldst, 0, 0);
#else
    *(uint4*)ldst = *(const uint4*)gsrc;
#endif
}

__device__ __forceinline__ void async_wait() {
#if HAVE_ASYNC_LDS
#if __has_builtin(__builtin_amdgcn_s_wait_asynccnt)
    __builtin_amdgcn_s_wait_asynccnt(0);
#else
    asm volatile("s_wait_asynccnt 0x0" ::: "memory");
#endif
#endif
}

// C[M,N] = act(A_bf16[M,K] @ W_f32[K,N] + bias) (+ residual), out f32 or bf16.
// act: 0 = none, 1 = exact-erf GELU
__global__ __launch_bounds__(256)
void gemm_bf16_wmma(const u16* __restrict__ A, const float* __restrict__ W,
                    const float* __restrict__ bias, const float* __restrict__ residual,
                    void* __restrict__ Cout, int M, int N, int K, int act, int out_bf16)
{
    __shared__ u16 sA[TM][TK + LDS_PAD];   // 18.4 KB
    __shared__ u16 sB[TN][TK + LDS_PAD];   // transposed [N][K]

    const int tid  = threadIdx.x;
    const int lane = tid & 31;
    const int wave = tid >> 5;
    const int m0 = blockIdx.y * TM;
    const int n0 = blockIdx.x * TN;

    // 8 waves: 4 stacked on M (32 rows), 2 on N (64 cols); wave tile 32x64
    const int wm = (wave & 3) * 32;
    const int wn = (wave >> 2) * 64;
    const int halfsel = lane >> 4;
    const int l16     = lane & 15;

    v8f acc[2][4];
    for (int i = 0; i < 2; ++i)
        for (int j = 0; j < 4; ++j)
            for (int r = 0; r < 8; ++r)
                acc[i][j][r] = 0.0f;

    for (int k0 = 0; k0 < K; k0 += TK) {
        // ---- A tile 128x64 bf16: pure copy, async to LDS (4 x b128/thread)
        for (int i = 0; i < 4; ++i) {
            const int f  = tid + i * 256;
            const int r  = f >> 3;          // 0..127
            const int c8 = (f & 7) << 3;    // 0..56, 8 bf16 per transfer
            async_copy16(A + (size_t)(m0 + r) * K + k0 + c8, &sA[r][c8]);
        }
        // ---- W tile 64x128 f32 -> bf16, transposed into [N][K] (8 float4/thread)
        for (int i = 0; i < 8; ++i) {
            const int f   = tid + i * 256;
            const int kr  = f >> 5;          // 0..63
            const int nc4 = (f & 31) << 2;   // 0..124
            const float4 v = *(const float4*)(W + (size_t)(k0 + kr) * N + n0 + nc4);
            sB[nc4 + 0][kr] = f32_to_bf16(v.x);
            sB[nc4 + 1][kr] = f32_to_bf16(v.y);
            sB[nc4 + 2][kr] = f32_to_bf16(v.z);
            sB[nc4 + 3][kr] = f32_to_bf16(v.w);
        }
        if (k0 + TK < K) {   // global_prefetch_b8 hints for next K tile
            __builtin_prefetch(A + (size_t)(m0 + (tid >> 1)) * K + k0 + TK, 0, 0);
            __builtin_prefetch(W + (size_t)(k0 + TK + (tid & 63)) * N + n0 + ((tid >> 6) << 5), 0, 0);
        }
        async_wait();
        __syncthreads();

        // ---- two WMMA K-steps per stage
        for (int kh = 0; kh < TK; kh += 32) {
            FragBF16 afr[2], bfr[4];
            for (int i = 0; i < 2; ++i)
                load_frag(&sA[wm + i * 16 + l16][kh + halfsel * 8], afr[i]);
            for (int j = 0; j < 4; ++j)
                load_frag(&sB[wn + j * 16 + l16][kh + halfsel * 8], bfr[j]);

            for (int i = 0; i < 2; ++i)
                for (int j = 0; j < 4; ++j)
                    acc[i][j] = __builtin_amdgcn_wmma_f32_16x16x32_bf16(
                        false, afr[i].v, false, bfr[j].v,
                        (short)0, acc[i][j], false, false);
        }
        __syncthreads();
    }

    // ---- epilogue: bias + optional GELU + optional residual
    // C/D layout: VGPR r -> M = r + 8*half ; lane%16 -> N
    float*  Cf = (float*)Cout;
    u16*    Ch = (u16*)Cout;
    for (int i = 0; i < 2; ++i) {
        const int mbase = m0 + wm + i * 16 + halfsel * 8;
        for (int j = 0; j < 4; ++j) {
            const int col  = n0 + wn + j * 16 + l16;
            const float bv = bias ? bias[col] : 0.0f;
            for (int r = 0; r < 8; ++r) {
                const size_t idx = (size_t)(mbase + r) * N + col;
                float v = acc[i][j][r] + bv;
                if (act == 1) v = 0.5f * v * (1.0f + erff(v * 0.70710678118f));
                if (residual) v += residual[idx];
                if (out_bf16) Ch[idx] = f32_to_bf16(v);
                else          Cf[idx] = v;
            }
        }
    }
}

// ---------------------------------------------------------------------------
__global__ __launch_bounds__(256)
void embed_kernel(const int* __restrict__ ids, const float* __restrict__ wte,
                  const float* __restrict__ wpe, float* __restrict__ x, int T, int D)
{
    const int bt = blockIdx.x;
    const int t  = bt % T;
    const int id = ids[bt];
    for (int d = threadIdx.x; d < D; d += blockDim.x)
        x[(size_t)bt * D + d] = wte[(size_t)id * D + d] + wpe[(size_t)t * D + d];
}

// D == 1024: 256 threads x 4 elements; f32 in, bf16 out (feeds WMMA A side)
__global__ __launch_bounds__(256)
void layernorm_bf16_kernel(const float* __restrict__ x, const float* __restrict__ g,
                           const float* __restrict__ b, u16* __restrict__ o, int D)
{
    __shared__ float redS[8], redS2[8];
    const int row = blockIdx.x;
    const float* xr = x + (size_t)row * D;
    float v[4], s = 0.0f, s2 = 0.0f;
    for (int i = 0; i < 4; ++i) {
        v[i] = xr[threadIdx.x + i * 256];
        s += v[i]; s2 += v[i] * v[i];
    }
    for (int off = 16; off; off >>= 1) {
        s  += __shfl_xor(s,  off, 32);
        s2 += __shfl_xor(s2, off, 32);
    }
    const int wv = threadIdx.x >> 5, lane = threadIdx.x & 31;
    if (lane == 0) { redS[wv] = s; redS2[wv] = s2; }
    __syncthreads();
    s = 0.0f; s2 = 0.0f;
    for (int i = 0; i < 8; ++i) { s += redS[i]; s2 += redS2[i]; }
    const float mean = s / D;
    const float rstd = rsqrtf(s2 / D - mean * mean + 1e-5f);
    for (int i = 0; i < 4; ++i) {
        const int d = threadIdx.x + i * 256;
        o[(size_t)row * D + d] = f32_to_bf16((v[i] - mean) * rstd * g[d] + b[d]);
    }
}

// Flash-style causal attention: 1 wave per query row, lane owns 2 of hd=64 dims.
// qkv is f32 [B,T,3D]; output y is bf16 [B,T,D] (feeds proj GEMM A side).
__global__ __launch_bounds__(256)
void attention_kernel(const float* __restrict__ qkv, u16* __restrict__ y,
                      int T, int D, int H)
{
    const int hd   = D / H;            // 64
    const int wave = threadIdx.x >> 5;
    const int lane = threadIdx.x & 31;
    const int q = blockIdx.x * 8 + wave;
    const int h = blockIdx.y;
    const int b = blockIdx.z;
    const size_t rs = 3 * (size_t)D;
    const float* base = qkv + (size_t)b * T * rs;
    const int d0 = lane * 2;

    const float2 qv = *(const float2*)(base + (size_t)q * rs + h * hd + d0);
    const float scale = rsqrtf((float)hd);

    float m = -INFINITY, l = 0.0f;
    float2 acc = make_float2(0.0f, 0.0f);
    for (int k = 0; k <= q; ++k) {
        const float2 kv = *(const float2*)(base + (size_t)k * rs + D + h * hd + d0);
        float s = qv.x * kv.x + qv.y * kv.y;
        for (int off = 16; off; off >>= 1) s += __shfl_xor(s, off, 32);
        s *= scale;
        const float mn   = fmaxf(m, s);
        const float corr = __expf(m - mn);      // first iter: exp(-inf)=0
        const float p    = __expf(s - mn);
        const float2 vv  = *(const float2*)(base + (size_t)k * rs + 2 * D + h * hd + d0);
        l     = l * corr + p;
        acc.x = acc.x * corr + p * vv.x;
        acc.y = acc.y * corr + p * vv.y;
        m = mn;
    }
    const float inv = 1.0f / l;
    u16* yr = y + ((size_t)b * T + q) * D + h * hd + d0;
    yr[0] = f32_to_bf16(acc.x * inv);
    yr[1] = f32_to_bf16(acc.y * inv);
}

// ---------------------------------------------------------------------------
static void launch_gemm(const u16* A, const float* W, const float* bias,
                        const float* residual, void* C, int M, int N, int K,
                        int act, int out_bf16, hipStream_t s)
{
    dim3 grid(N / TN, M / TM), block(256);
    hipLaunchKernelGGL(gemm_bf16_wmma, grid, block, 0, s,
                       A, W, bias, residual, C, M, N, K, act, out_bf16);
}

extern "C" void kernel_launch(void* const* d_in, const int* in_sizes, int n_in,
                              void* d_out, int out_size, void* d_ws, size_t ws_size,
                              hipStream_t stream)
{
    constexpr int B = 2, T = 1024, D = 1024, L = 4, H = 16, V = 32000;
    const int M = B * T;  // 2048

    const int*   ids    = (const int*)  d_in[0];
    const float* wte    = (const float*)d_in[1];
    const float* wpe    = (const float*)d_in[2];
    const float* ln1_g  = (const float*)d_in[3];
    const float* ln1_b  = (const float*)d_in[4];
    const float* w_qkv  = (const float*)d_in[5];
    const float* b_qkv  = (const float*)d_in[6];
    const float* w_proj = (const float*)d_in[7];
    const float* b_proj = (const float*)d_in[8];
    const float* ln2_g  = (const float*)d_in[9];
    const float* ln2_b  = (const float*)d_in[10];
    const float* w_fc1  = (const float*)d_in[11];
    const float* b_fc1  = (const float*)d_in[12];
    const float* w_fc2  = (const float*)d_in[13];
    const float* b_fc2  = (const float*)d_in[14];
    const float* lnf_g  = (const float*)d_in[15];
    const float* lnf_b  = (const float*)d_in[16];
    const float* w_lm   = (const float*)d_in[17];
    float* out = (float*)d_out;

    // workspace: x f32 | a bf16 | y bf16 | qkv f32 | h2 bf16  (~59 MB)
    char* w = (char*)d_ws;
    const size_t S = (size_t)M * D;                  // 2,097,152 elements
    float* x    = (float*)(w);                       // [M,D]  f32 residual stream
    u16*   a    = (u16*)  (w + S * 4);               // [M,D]  bf16 LN output
    u16*   y    = (u16*)  (w + S * 6);               // [M,D]  bf16 attention out
    float* qkvb = (float*)(w + S * 8);               // [M,3D] f32
    u16*   h2   = (u16*)  (w + S * 20);              // [M,4D] bf16 MLP hidden

    hipLaunchKernelGGL(embed_kernel, dim3(M), dim3(256), 0, stream,
                       ids, wte, wpe, x, T, D);

    for (int l = 0; l < L; ++l) {
        const float* Wq = w_qkv  + (size_t)l * D * 3 * D;
        const float* bq = b_qkv  + (size_t)l * 3 * D;
        const float* Wp = w_proj + (size_t)l * D * D;
        const float* bp = b_proj + (size_t)l * D;
        const float* W1 = w_fc1  + (size_t)l * D * 4 * D;
        const float* b1 = b_fc1  + (size_t)l * 4 * D;
        const float* W2 = w_fc2  + (size_t)l * 4 * D * D;
        const float* b2 = b_fc2  + (size_t)l * D;

        // a = LN1(x); qkv = a @ Wq + bq
        hipLaunchKernelGGL(layernorm_bf16_kernel, dim3(M), dim3(256), 0, stream,
                           x, ln1_g + (size_t)l * D, ln1_b + (size_t)l * D, a, D);
        launch_gemm(a, Wq, bq, nullptr, qkvb, M, 3 * D, D, 0, 0, stream);

        // y = causal attention(qkv)
        hipLaunchKernelGGL(attention_kernel, dim3(T / 8, H, B), dim3(256), 0, stream,
                           qkvb, y, T, D, H);

        // x = x + y @ Wp + bp
        launch_gemm(y, Wp, bp, x, x, M, D, D, 0, 0, stream);

        // a = LN2(x); h2 = gelu(a @ W1 + b1) [bf16]; x = x + h2 @ W2 + b2
        hipLaunchKernelGGL(layernorm_bf16_kernel, dim3(M), dim3(256), 0, stream,
                           x, ln2_g + (size_t)l * D, ln2_b + (size_t)l * D, a, D);
        launch_gemm(a, W1, b1, nullptr, h2, M, 4 * D, D, 1, 1, stream);
        launch_gemm(h2, W2, b2, x, x, M, D, 4 * D, 0, 0, stream);
    }

    // logits = LN_f(x) @ w_lm  (no bias)
    hipLaunchKernelGGL(layernorm_bf16_kernel, dim3(M), dim3(256), 0, stream,
                       x, lnf_g, lnf_b, a, D);
    launch_gemm(a, w_lm, nullptr, nullptr, out, M, V, D, 0, 0, stream);
}
